// Decoder_987842478702
// MI455X (gfx1250) — matrix operand order repaired and checked by reading, not compiled, not run on hardware
//
#include <hip/hip_runtime.h>
#include <hip/hip_bf16.h>

// ---------------------------------------------------------------------------
// Decoder: concat/mask -> GRU(512) -> GRU(512) -> Dense(1, tanh) * mask
// B=128, T=512, LAT=256, F2=64, D1=320, H=512
//
// MI455X reasoning: ~3.7e11 bf16 FLOPs and ~170MB of activation traffic are
// both negligible (23.3 TB/s HBM, 192MB L2 holds all weights). The cost is the
// T=512 serial scan. So: ONE persistent kernel runs the whole scan — 32 WGs
// compute GRU-5 step t while 32 WGs compute GRU-6 step t-1 (software
// pipeline), separated by an atomic device-wide barrier. Each wave owns a
// 16x16 h-tile and fuses the 3 gate GEMMs (x@W on the fly + h@U) into
// v_wmma_f32_16x16x32_bf16 chains with gate nonlinearities, reading
// pre-transposed bf16 weights as contiguous 16B loads.
// ---------------------------------------------------------------------------

typedef __attribute__((ext_vector_type(16))) __bf16 v16bf;
typedef __attribute__((ext_vector_type(8)))  float  v8f;

#define GRU_H 512
#define SCAN_BLOCKS 64u

__device__ __forceinline__ float hsig(float x) {
    return fminf(fmaxf(0.2f * x + 0.5f, 0.0f), 1.0f);
}

// A-operand: 16x32 bf16 tile (ISA 16-bit A layout):
//   lanes 0-15 : M=lane,    halves 0-7 -> K 0..7,  8-15 -> K 16..23
//   lanes 16-31: M=lane-16, halves 0-7 -> K 8..15, 8-15 -> K 24..31
__device__ __forceinline__ v16bf load_tile_A(const __bf16* __restrict__ base,
                                             size_t ld, int lane) {
    int m  = lane & 15;
    int kb = (lane >> 4) << 3;               // 0 or 8
    const __bf16* p = base + (size_t)m * ld + kb;
    union { uint4 u[2]; v16bf v; } t;
    t.u[0] = *reinterpret_cast<const uint4*>(p);        // K = kb .. kb+7
    t.u[1] = *reinterpret_cast<const uint4*>(p + 16);   // K = kb+16 .. kb+23
    return t.v;
}

// B-operand: 32x16 bf16 tile from pre-transposed weights Wt[N_total][Kd]
// (row = output column, contiguous along K). ISA B 32x16 layout:
//   lanes 0-15 : N = row0+lane,    K = k0 .. k0+7,  k0+16 .. k0+23
//   lanes 16-31: N = row0+lane-16, K = k0+8 .. k0+15, k0+24 .. k0+31
__device__ __forceinline__ v16bf load_tile_B(const __bf16* __restrict__ wt,
                                             size_t kd, int row0, int k0,
                                             int lane) {
    int nn = lane & 15;
    int kb = (lane >> 4) << 3;               // 0 or 8
    const __bf16* p = wt + (size_t)(row0 + nn) * kd + (size_t)(k0 + kb);
    union { uint4 u[2]; v16bf v; } t;
    t.u[0] = *reinterpret_cast<const uint4*>(p);
    t.u[1] = *reinterpret_cast<const uint4*>(p + 16);
    return t.v;
}

__device__ __forceinline__ v8f wmma_bf16(v16bf a, v16bf b, v8f c) {
    return __builtin_amdgcn_wmma_f32_16x16x32_bf16(
        /*neg_a=*/false, a, /*neg_b=*/false, b,
        /*c_mod=*/(short)0, c, /*reuse_a=*/false, /*reuse_b=*/false);
}

// One 16x16 tile of one GRU timestep: 3 gate GEMM tiles + gate math.
__device__ __forceinline__ void gru_tile_step(
    const __bf16* __restrict__ x_t, size_t x_stride, int D,
    const __bf16* __restrict__ Wt,     // [3H, D]  bf16, transposed
    const float*  __restrict__ bi,     // [3H]
    const __bf16* __restrict__ h_prev, size_t h_stride,
    const __bf16* __restrict__ Ut,     // [3H, H]  bf16, transposed
    const float*  __restrict__ br,     // [3H]
    __bf16* __restrict__ h_out, size_t o_stride,
    int tile, int lane)
{
    const int H = GRU_H;
    const int n0 = (tile & 31) * 16;          // hidden column block
    const int m0 = (tile >> 5) * 16;          // batch row block

    v8f acc_z = {}, acc_r = {}, acc_xh = {}, acc_ih = {};

    // Input projection: x_t[m0..m0+15, :] @ W
    const __bf16* xrow = x_t + (size_t)m0 * x_stride;
    for (int k0 = 0; k0 < D; k0 += 32) {
        v16bf a = load_tile_A(xrow + k0, x_stride, lane);
        acc_z  = wmma_bf16(a, load_tile_B(Wt, (size_t)D, n0,         k0, lane), acc_z);
        acc_r  = wmma_bf16(a, load_tile_B(Wt, (size_t)D, H + n0,     k0, lane), acc_r);
        acc_xh = wmma_bf16(a, load_tile_B(Wt, (size_t)D, 2 * H + n0, k0, lane), acc_xh);
    }

    // Recurrent projection: h_prev @ U
    const __bf16* hrow = h_prev + (size_t)m0 * h_stride;
    for (int k0 = 0; k0 < H; k0 += 32) {
        v16bf a = load_tile_A(hrow + k0, h_stride, lane);
        acc_z  = wmma_bf16(a, load_tile_B(Ut, (size_t)H, n0,         k0, lane), acc_z);
        acc_r  = wmma_bf16(a, load_tile_B(Ut, (size_t)H, H + n0,     k0, lane), acc_r);
        acc_ih = wmma_bf16(a, load_tile_B(Ut, (size_t)H, 2 * H + n0, k0, lane), acc_ih);
    }

    // Gate math. C layout: VGPR i -> (M = m0 + (lane>=16)*8 + i, N = n0 + lane&15)
    const int n  = n0 + (lane & 15);
    const int mo = m0 + ((lane >> 4) << 3);
    const float bz  = bi[n]         + br[n];
    const float brg = bi[H + n]     + br[H + n];
    const float bxh = bi[2 * H + n];
    const float bih = br[2 * H + n];
#pragma unroll
    for (int i = 0; i < 8; ++i) {
        const int m = mo + i;
        float hp = (float)h_prev[(size_t)m * h_stride + n];
        float zg = hsig(acc_z[i] + bz);
        float rg = hsig(acc_r[i] + brg);
        float hh = tanhf(acc_xh[i] + bxh + rg * (acc_ih[i] + bih));
        float hn = zg * hp + (1.0f - zg) * hh;
        h_out[(size_t)m * o_stride + n] = (__bf16)hn;
    }
}

// Sense-free generation-count device barrier (all SCAN_BLOCKS co-resident).
__device__ __forceinline__ void grid_barrier(unsigned* cnt,
                                             volatile unsigned* gen) {
    __threadfence();                 // make this block's h writes visible
    __syncthreads();
    if (threadIdx.x == 0) {
        unsigned g = *gen;
        unsigned old = atomicAdd(cnt, 1u);
        if (old == SCAN_BLOCKS - 1u) {
            *(volatile unsigned*)cnt = 0u;      // all arrived; reset for reuse
            __threadfence();
            atomicAdd((unsigned*)gen, 1u);      // release everyone
        } else {
            while (*gen == g) { __builtin_amdgcn_s_sleep(2); }
        }
    }
    __syncthreads();
    __threadfence();                 // acquire: invalidate stale L0 lines
}

// ---------------------------------------------------------------------------
// Persistent scan: WGs 0..31 run GRU-5 (step t), WGs 32..63 run GRU-6
// (step t-1), barrier per iteration. T+1 iterations drain the pipeline.
// ---------------------------------------------------------------------------
__global__ __launch_bounds__(256) void gru_scan_kernel(
    const __bf16* __restrict__ X1,    // [B,T,D1] bf16 masked concat
    const __bf16* __restrict__ Wt5, const float* __restrict__ bi5,
    const __bf16* __restrict__ Ut5, const float* __restrict__ br5,
    const __bf16* __restrict__ Wt6, const float* __restrict__ bi6,
    const __bf16* __restrict__ Ut6, const float* __restrict__ br6,
    const __bf16* __restrict__ h0,    // [B,H] zeros
    __bf16* __restrict__ g5,          // [B,T,H]
    __bf16* __restrict__ g6,          // [B,T,H]
    unsigned* __restrict__ bar)       // bar[0]=cnt, bar[1]=gen
{
    const int T = 512, H = GRU_H, D1 = 320;
    const size_t sBT = (size_t)T * H;

    const int lane = threadIdx.x & 31;
    const int wave = threadIdx.x >> 5;
    const bool layer6 = (blockIdx.x >= 32);
    const int tile = ((blockIdx.x & 31) * 8) + wave;   // 0..255

    for (int it = 0; it <= T; ++it) {
        if (!layer6) {
            if (it < T) {
                const __bf16* hp = (it == 0) ? h0 : g5 + (size_t)(it - 1) * H;
                size_t hs        = (it == 0) ? (size_t)H : sBT;
                gru_tile_step(X1 + (size_t)it * D1, (size_t)T * D1, D1,
                              Wt5, bi5, hp, hs, Ut5, br5,
                              g5 + (size_t)it * H, sBT, tile, lane);
            }
        } else {
            if (it >= 1) {
                const int t = it - 1;
                const __bf16* hp = (t == 0) ? h0 : g6 + (size_t)(t - 1) * H;
                size_t hs        = (t == 0) ? (size_t)H : sBT;
                gru_tile_step(g5 + (size_t)t * H, sBT, H,
                              Wt6, bi6, hp, hs, Ut6, br6,
                              g6 + (size_t)t * H, sBT, tile, lane);
            }
        }
        if (it < T) grid_barrier(bar, bar + 1);
    }
}

// ---------------------------------------------------------------------------
// Build masked concat input in bf16: X1[b,t,d] = concat(z, tin2)[d] * mask[b,t]
__global__ __launch_bounds__(256) void build_x_kernel(
    const float* __restrict__ z, const float* __restrict__ tin2,
    const float* __restrict__ masks, __bf16* __restrict__ X1, int total)
{
    const int T = 512, D = 320, LAT = 256, F2 = 64;
    int idx = blockIdx.x * 256 + threadIdx.x;
    if (idx >= total) return;
    int d = idx % D;
    int bt = idx / D;
    int b = bt / T;
    float v = (d < LAT) ? z[b * LAT + d] : tin2[(size_t)bt * F2 + (d - LAT)];
    X1[idx] = (__bf16)(v * masks[bt]);
}

// Transpose + convert: Wt[n,k] = (bf16) W[k,n];  W: [K,N] fp32, Wt: [N,K] bf16
__global__ __launch_bounds__(256) void transpose_w_kernel(
    const float* __restrict__ W, __bf16* __restrict__ Wt, int K, int N)
{
    int idx = blockIdx.x * 256 + threadIdx.x;
    if (idx >= K * N) return;
    int nn = idx / K;
    int k  = idx % K;
    Wt[idx] = (__bf16)W[(size_t)k * N + nn];
}

// Zero h0 + barrier state (deterministic per call).
__global__ __launch_bounds__(256) void init_state_kernel(
    __bf16* __restrict__ h0, int n, unsigned* __restrict__ bar)
{
    int i = blockIdx.x * 256 + threadIdx.x;
    if (i < n) h0[i] = (__bf16)0.0f;
    if (i == 0) { bar[0] = 0u; bar[1] = 0u; }
}

// Final TimeDistributed Dense(1, tanh) * dec_masks. One wave per (b,t).
__global__ __launch_bounds__(256) void dense_tanh_kernel(
    const __bf16* __restrict__ g6,   // [B*T, H]
    const float*  __restrict__ Wd,   // [H]
    const float*  __restrict__ bd,   // [1]
    const float*  __restrict__ dmask,// [B*T]
    float* __restrict__ out, int total)
{
    const int H = GRU_H;
    int lane = threadIdx.x & 31;
    int row  = (blockIdx.x * 256 + threadIdx.x) >> 5;
    if (row >= total) return;
    const __bf16* g = g6 + (size_t)row * H;
    float s = 0.0f;
    for (int h = lane; h < H; h += 32) s += (float)g[h] * Wd[h];
#pragma unroll
    for (int off = 16; off; off >>= 1) s += __shfl_xor(s, off, 32);
    if (lane == 0) out[row] = tanhf(s + bd[0]) * dmask[row];
}

// ---------------------------------------------------------------------------
extern "C" void kernel_launch(void* const* d_in, const int* in_sizes, int n_in,
                              void* d_out, int out_size, void* d_ws, size_t ws_size,
                              hipStream_t stream)
{
    const int B = 128, T = 512, LAT = 256, F2 = 64, H = GRU_H;
    const int D1 = LAT + F2;                 // 320

    const float* z    = (const float*)d_in[0];
    const float* tin2 = (const float*)d_in[1];
    const float* mask = (const float*)d_in[2];
    const float* dmsk = (const float*)d_in[3];
    const float* W5   = (const float*)d_in[4];
    const float* U5   = (const float*)d_in[5];
    const float* bi5  = (const float*)d_in[6];
    const float* br5  = (const float*)d_in[7];
    const float* W6   = (const float*)d_in[8];
    const float* U6   = (const float*)d_in[9];
    const float* bi6  = (const float*)d_in[10];
    const float* br6  = (const float*)d_in[11];
    const float* Wd   = (const float*)d_in[12];
    const float* bd   = (const float*)d_in[13];
    float* out = (float*)d_out;

    // Workspace carve-out (~175 MB, 256B aligned slices)
    char* ws = (char*)d_ws;
    auto carve = [&](size_t bytes) {
        char* p = ws;
        ws += (bytes + 255) & ~(size_t)255;
        return p;
    };
    __bf16* X1  = (__bf16*)carve((size_t)B * T * D1 * 2);   // masked concat
    __bf16* Wt5 = (__bf16*)carve((size_t)3 * H * D1 * 2);
    __bf16* Ut5 = (__bf16*)carve((size_t)3 * H * H  * 2);
    __bf16* Wt6 = (__bf16*)carve((size_t)3 * H * H  * 2);
    __bf16* Ut6 = (__bf16*)carve((size_t)3 * H * H  * 2);
    __bf16* g5  = (__bf16*)carve((size_t)B * T * H  * 2);   // GRU5 outputs (= h states)
    __bf16* g6  = (__bf16*)carve((size_t)B * T * H  * 2);   // GRU6 outputs
    __bf16* h0  = (__bf16*)carve((size_t)B * H * 2);        // zero initial state
    unsigned* bar = (unsigned*)carve(256);                  // barrier cnt/gen

    // --- Preparation (fully parallel) ---
    {
        int tot = B * T * D1;
        build_x_kernel<<<(tot + 255) / 256, 256, 0, stream>>>(z, tin2, mask, X1, tot);
    }
    transpose_w_kernel<<<(D1 * 3 * H + 255) / 256, 256, 0, stream>>>(W5, Wt5, D1, 3 * H);
    transpose_w_kernel<<<(H  * 3 * H + 255) / 256, 256, 0, stream>>>(U5, Ut5, H,  3 * H);
    transpose_w_kernel<<<(H  * 3 * H + 255) / 256, 256, 0, stream>>>(W6, Wt6, H,  3 * H);
    transpose_w_kernel<<<(H  * 3 * H + 255) / 256, 256, 0, stream>>>(U6, Ut6, H,  3 * H);
    init_state_kernel<<<(B * H + 255) / 256, 256, 0, stream>>>(h0, B * H, bar);

    // --- Persistent pipelined scan over all T steps, both layers ---
    gru_scan_kernel<<<SCAN_BLOCKS, 256, 0, stream>>>(
        X1, Wt5, bi5, Ut5, br5, Wt6, bi6, Ut6, br6, h0, g5, g6, bar);

    // --- Output head ---
    {
        int tot = B * T;                    // 65536 rows, one wave each
        dense_tanh_kernel<<<(tot * 32 + 255) / 256, 256, 0, stream>>>(
            g6, Wd, bd, dmsk, out, tot);
    }
}
